// DLPTDownLayer_74586402062450
// MI455X (gfx1250) — matrix operand order repaired and checked by reading, not compiled, not run on hardware
//
#include <hip/hip_runtime.h>
#include <math.h>

// ---------------------------------------------------------------------------
// Types for CDNA5 WMMA (gfx1250, wave32):  D(f32 16x16) = A(f16 16x32) x B(f16 32x16) + C
// ---------------------------------------------------------------------------
typedef _Float16 f16;
typedef __attribute__((ext_vector_type(16))) _Float16      f16x16;
typedef __attribute__((ext_vector_type(8)))  float         f32x8;
typedef __attribute__((ext_vector_type(4)))  unsigned int  u32x4;

union Frag { f16x16 v; u32x4 q[2]; };

static __device__ __forceinline__ f32x8 wmma_step(const Frag& a, const Frag& b, f32x8 c) {
  return __builtin_amdgcn_wmma_f32_16x16x32_f16(false, a.v, false, b.v, (short)0, c, false, false);
}

// LDS strides (in halves / floats), padded to dodge bank conflicts, all 16B aligned
constexpr int kLDA = 264;   // 64x256 f16 activation buffers
constexpr int kLDV = 72;    // vT [256][64] and probs [64][64] f16 buffers
constexpr int kLDS = 68;    // scores [64][64] f32

constexpr size_t SMEM_BYTES =
    (size_t)(5 * 64 * kLDA) * 2   // sX, sHpos, sHgeo, sQ, sK
  + (size_t)(256 * kLDV) * 2      // sVT
  + (size_t)(64 * kLDS) * 4       // sS (scores, f32)
  + (size_t)(64 * kLDV) * 2      // sP (probs, f16)
  + (size_t)(64 * 4 + 64 + 12 + 12) * 4;  // lp, nrm, cog, avg

// ---------------------------------------------------------------------------
// Shared 64-row GEMM: OUT[64 x (NCT*16)] = A[64 x (KS*32)] @ Bt^T, WMMA f16->f32.
// A is LDS f16 row-major (lda halves); Bt is [N][K] f16 row-major (ldb halves).
// Epilogue selected at compile time (BIAS / RG global-f32 resid / RL lds-f16 resid
// / TRANS packed transposed store) so the hot loop carries no runtime branches;
// bias+resid loads are issued BEFORE the WMMA chain to overlap with B traffic.
// ---------------------------------------------------------------------------
template <int KS, int NCT, bool BIAS, bool RG, bool RL, bool TRANS>
static __device__ __forceinline__ void gemm64(
    const f16* __restrict__ sA, int lda,
    const f16* __restrict__ Bt, int ldb,
    float scale, const float* __restrict__ bias,
    const float* __restrict__ residG,
    const f16* __restrict__ residL, int ldr,
    f16* __restrict__ outp, int ldo)
{
  const int tid  = threadIdx.x;
  const int w    = tid >> 5, lane = tid & 31;
  const int kh   = lane >> 4, ln = lane & 15;   // half-wave, sub-lane
  const int mt   = w & 3, wg = w >> 2;
  const int M0   = mt * 16;

  // A fragments for this wave's 16 rows, all K-steps, kept resident
  Frag a[KS];
#pragma unroll
  for (int s = 0; s < KS; ++s) {
    const f16* p = sA + (M0 + ln) * lda + s * 32 + kh * 8;
    a[s].q[0] = *(const u32x4*)p;
    a[s].q[1] = *(const u32x4*)(p + 16);
  }

#pragma unroll 1
  for (int j = 0; j < NCT / 2; ++j) {
    const int nt = wg + 2 * j;
    const int N  = nt * 16 + ln;

    // epilogue operands first: overlap their latency with the WMMA chain
    float bi = 0.0f;
    if (BIAS) bi = bias[N];
    float rg[8];
    if (RG) {
#pragma unroll
      for (int r = 0; r < 8; ++r) rg[r] = residG[(size_t)(M0 + kh * 8 + r) * 256 + N];
    }
    float rl[8];
    if (RL) {
#pragma unroll
      for (int r = 0; r < 8; ++r) rl[r] = (float)residL[(M0 + kh * 8 + r) * ldr + N];
    }

    f32x8 acc = {};
    const f16* bp = Bt + (size_t)N * ldb + kh * 8;
#pragma unroll
    for (int s = 0; s < KS; ++s) {
      Frag b;
      b.q[0] = *(const u32x4*)bp;
      b.q[1] = *(const u32x4*)(bp + 16);
      bp += 32;
      acc = wmma_step(a[s], b, acc);
    }

    if (TRANS) {
      union { f16 h[8]; u32x4 q; } pk;
#pragma unroll
      for (int r = 0; r < 8; ++r) pk.h[r] = (f16)(acc[r] * scale + bi);
      *(u32x4*)(outp + (size_t)N * ldo + M0 + kh * 8) = pk.q;
    } else {
#pragma unroll
      for (int r = 0; r < 8; ++r) {
        const int M = M0 + kh * 8 + r;
        float v = acc[r] * scale + bi;
        if (RG) v += rg[r];
        if (RL) v += rl[r];
        outp[M * ldo + N] = (f16)v;
      }
    }
  }
}

// ---------------------------------------------------------------------------
// Weight transpose + f32 -> f16:  dst[n*K + k] = (f16) src[k*N + n]
// ---------------------------------------------------------------------------
__global__ void convT_kernel(const float* __restrict__ src, f16* __restrict__ dst,
                             int K, int N) {
  int i = blockIdx.x * 256 + threadIdx.x;
  if (i < K * N) {
    int n = i / K, k = i - n * K;
    dst[i] = (f16)src[(size_t)k * N + n];
  }
}

// ---------------------------------------------------------------------------
// One local-attention block over a 64-row tile (covers 4 chunks of 16 or 1 of 64)
// ---------------------------------------------------------------------------
struct BlockArgs {
  const float* pos; const float* fin; float* fout;
  const f16 *m1w2t, *m2w2t, *wqt, *wkt, *wvt, *wot;     // [256][256] f16 (N-major)
  const float *m1w1, *m1b1, *m1b2, *m2w1, *m2b1, *m2b2; // small fp32
  const float *bo, *lng, *lnb;
  int cs;
};

__global__ __launch_bounds__(256) void block_kernel(BlockArgs A) {
  extern __shared__ __align__(16) unsigned char smem_dyn[];
  f16*   sX   = (f16*)smem_dyn;            // [64][kLDA]  staging (t1/t2/attn-out)
  f16*   sHp  = sX  + 64 * kLDA;           // h_pos  (f16)
  f16*   sHg  = sHp + 64 * kLDA;           // h_geo, later y_pre-LN
  f16*   sQ   = sHg + 64 * kLDA;
  f16*   sK   = sQ  + 64 * kLDA;
  f16*   sVT  = sK  + 64 * kLDA;           // [256][kLDV]  v transposed
  float* sS   = (float*)(sVT + 256 * kLDV);// [64][kLDS]   scores f32
  f16*   sP   = (f16*)(sS + 64 * kLDS);    // [64][kLDV]   probs f16 (zero off-block)
  float* sLP  = (float*)(sP + 64 * kLDV);  // [64][4] lp (xyz, pad)
  float* sNrm = sLP + 64 * 4;              // [64]
  float* sCog = sNrm + 64;                 // [4][3]
  float* sAvg = sCog + 12;                 // [4][3]

  const int tid  = threadIdx.x;
  const int w    = tid >> 5, lane = tid & 31;
  const int kh   = lane >> 4, ln = lane & 15;
  const int rowBase = blockIdx.x * 64;     // flat global row (B*N flattened)
  const int cs = A.cs, nch = 64 / cs;

  // ---- geometry: cog, lp, nrm, avg(≈0) ---------------------------------
  if (tid < 64) {
    const float* pp = A.pos + (size_t)(rowBase + tid) * 3;
    sLP[tid * 4 + 0] = pp[0]; sLP[tid * 4 + 1] = pp[1]; sLP[tid * 4 + 2] = pp[2];
  }
  __syncthreads();
  if (tid < nch * 3) {
    int c = tid / 3, dd = tid - c * 3;
    float s = 0.f;
    for (int i2 = 0; i2 < cs; ++i2) s += sLP[(c * cs + i2) * 4 + dd];
    sCog[c * 3 + dd] = s / (float)cs;
  }
  __syncthreads();
  if (tid < 64) {
    int c = tid / cs;
    float l0 = sLP[tid * 4 + 0] - sCog[c * 3 + 0];
    float l1 = sLP[tid * 4 + 1] - sCog[c * 3 + 1];
    float l2 = sLP[tid * 4 + 2] - sCog[c * 3 + 2];
    sLP[tid * 4 + 0] = l0; sLP[tid * 4 + 1] = l1; sLP[tid * 4 + 2] = l2;
    sNrm[tid] = sqrtf(l0 * l0 + l1 * l1 + l2 * l2);
  }
  __syncthreads();
  if (tid < nch * 3) {
    int c = tid / 3, dd = tid - c * 3;
    float s = 0.f;
    for (int i2 = 0; i2 < cs; ++i2) s += sLP[(c * cs + i2) * 4 + dd];
    sAvg[c * 3 + dd] = s / (float)cs;
  }
  __syncthreads();

  // ---- MLP1 layer1: t1 = relu([lp,nrm] @ w1 + b1)  (K=4, VALU) ----------
  {
    const int c = tid;
    float w0 = A.m1w1[c], w1 = A.m1w1[256 + c], w2 = A.m1w1[512 + c], w3 = A.m1w1[768 + c];
    float bb = A.m1b1[c];
    for (int r = 0; r < 64; ++r) {
      float v = bb + sLP[r * 4 + 0] * w0 + sLP[r * 4 + 1] * w1 + sLP[r * 4 + 2] * w2
                   + sNrm[r] * w3;
      sX[r * kLDA + c] = (f16)fmaxf(v, 0.f);
    }
  }
  __syncthreads();
  // h_pos = fc + t1 @ m1w2 + b2
  gemm64<8, 16, true, true, false, false>(sX, kLDA, A.m1w2t, 256, 1.f, A.m1b2,
                A.fin + (size_t)rowBase * 256, nullptr, 0, sHp, kLDA);
  __syncthreads();

  // ---- MLP2 layer1: t2 = relu([avg,lp] @ w1 + b1)  (K=6, VALU) ----------
  {
    const int c = tid;
    float w0 = A.m2w1[c],        w1 = A.m2w1[256 + c],  w2 = A.m2w1[512 + c];
    float w3 = A.m2w1[768 + c],  w4 = A.m2w1[1024 + c], w5 = A.m2w1[1280 + c];
    float bb = A.m2b1[c];
    for (int r = 0; r < 64; ++r) {
      int ch = r / cs;
      float v = bb + sAvg[ch * 3 + 0] * w0 + sAvg[ch * 3 + 1] * w1 + sAvg[ch * 3 + 2] * w2
                   + sLP[r * 4 + 0] * w3 + sLP[r * 4 + 1] * w4 + sLP[r * 4 + 2] * w5;
      sX[r * kLDA + c] = (f16)fmaxf(v, 0.f);
    }
  }
  __syncthreads();
  // h_geo = fc + t2 @ m2w2 + b2
  gemm64<8, 16, true, true, false, false>(sX, kLDA, A.m2w2t, 256, 1.f, A.m2b2,
                A.fin + (size_t)rowBase * 256, nullptr, 0, sHg, kLDA);
  __syncthreads();

  // ---- Q (scaled 1/sqrt(256)), K, V (transposed into sVT) ---------------
  gemm64<8, 16, false, false, false, false>(sHg, kLDA, A.wqt, 256, 0.0625f, nullptr,
                nullptr, nullptr, 0, sQ, kLDA);
  gemm64<8, 16, false, false, false, false>(sHg, kLDA, A.wkt, 256, 1.f, nullptr,
                nullptr, nullptr, 0, sK, kLDA);
  gemm64<8, 16, false, false, false, true>(sHp, kLDA, A.wvt, 256, 1.f, nullptr,
                nullptr, nullptr, 0, sVT, kLDV);
  {  // zero probs buffer (off-diagonal blocks must be 0 for cs==16)
    unsigned int* z = (unsigned int*)sP;
    for (int i2 = tid; i2 < 64 * kLDV / 2; i2 += 256) z[i2] = 0u;
  }
  __syncthreads();

  // ---- scores: S = Q @ K^T within chunks (block-diagonal tiles) ---------
  {
    const int totalT = (cs == 64) ? 16 : 4;
    for (int T = w; T < totalT; T += 8) {   // wave-uniform: EXEC stays all-ones
      int mt2 = (cs == 64) ? (T >> 2) : T;
      int nt2 = (cs == 64) ? (T & 3)  : T;
      Frag a2[8];
#pragma unroll
      for (int s = 0; s < 8; ++s) {
        const f16* p = sQ + (mt2 * 16 + ln) * kLDA + s * 32 + kh * 8;
        a2[s].q[0] = *(const u32x4*)p;
        a2[s].q[1] = *(const u32x4*)(p + 16);
      }
      f32x8 acc = {};
      const f16* bp = sK + (nt2 * 16 + ln) * kLDA + kh * 8;
#pragma unroll
      for (int s = 0; s < 8; ++s) {
        Frag b2;
        b2.q[0] = *(const u32x4*)bp;
        b2.q[1] = *(const u32x4*)(bp + 16);
        bp += 32;
        acc = wmma_step(a2[s], b2, acc);
      }
#pragma unroll
      for (int r = 0; r < 8; ++r)
        sS[(mt2 * 16 + kh * 8 + r) * kLDS + nt2 * 16 + ln] = acc[r];
    }
  }
  __syncthreads();

  // ---- softmax (f32, per row within its chunk) --------------------------
  if (tid < 64) {
    const int base = (tid / cs) * cs;
    float mx = -3.0e38f;
    for (int j2 = 0; j2 < cs; ++j2) mx = fmaxf(mx, sS[tid * kLDS + base + j2]);
    float sum = 0.f;
    for (int j2 = 0; j2 < cs; ++j2) {
      float e = __expf(sS[tid * kLDS + base + j2] - mx);
      sum += e;
      sS[tid * kLDS + base + j2] = e;
    }
    float inv = 1.0f / sum;
    for (int j2 = 0; j2 < cs; ++j2)
      sP[tid * kLDV + base + j2] = (f16)(sS[tid * kLDS + base + j2] * inv);
  }
  __syncthreads();

  // ---- attn_out = P @ V  (K = 64 keys, V already transposed) ------------
  gemm64<2, 16, false, false, false, false>(sP, kLDV, sVT, kLDV, 1.f, nullptr,
                nullptr, nullptr, 0, sX, kLDA);
  __syncthreads();

  // ---- y_pre = h_pos + attn_out @ wo + bo -------------------------------
  gemm64<8, 16, true, false, true, false>(sX, kLDA, A.wot, 256, 1.f, A.bo,
                nullptr, sHp, kLDA, sHg, kLDA);
  __syncthreads();

  // ---- LayerNorm + write f_out (f32) ------------------------------------
  {
    const int r = tid >> 2, seg = tid & 3;
    const f16* row = sHg + r * kLDA + seg * 64;
    float s = 0.f, ss = 0.f;
    for (int j2 = 0; j2 < 64; ++j2) { float v = (float)row[j2]; s += v; ss += v * v; }
    s  += __shfl_xor(s, 1);  s  += __shfl_xor(s, 2);
    ss += __shfl_xor(ss, 1); ss += __shfl_xor(ss, 2);
    const float mu  = s * (1.f / 256.f);
    const float var = ss * (1.f / 256.f) - mu * mu;
    const float rs  = rsqrtf(var + 1e-5f);
    float* o = A.fout + (size_t)(rowBase + r) * 256 + seg * 64;
    for (int j2 = 0; j2 < 64; ++j2) {
      const int cI = seg * 64 + j2;
      o[j2] = ((float)row[j2] - mu) * rs * A.lng[cI] + A.lnb[cI];
    }
  }
}

// ---------------------------------------------------------------------------
// Fused transition-down (256->512) + BN + ReLU + KNN gather + max(16)
// One WG = 4 output points x 16 neighbors = 64 gathered rows
// ---------------------------------------------------------------------------
struct DownArgs {
  const float* f2; const int* knn; const f16* tdwT;   // tdwT: [512][256] f16
  const float *tdb, *bng, *bnb, *bnm, *bnv;
  float* outF;                                        // feat_ds region of d_out
};

__global__ __launch_bounds__(256) void down_kernel(DownArgs A) {
  __shared__ f16 sX[64 * kLDA];
  __shared__ int sIdx[64];
  const int tid = threadIdx.x;
  const int b   = blockIdx.x >> 9;          // 512 WGs per batch (2048/4)
  const int p0  = (blockIdx.x & 511) * 4;

  if (tid < 64) {
    int p = tid >> 4, k2 = tid & 15;
    sIdx[tid] = A.knn[((size_t)(b * 2048 + p0 + p)) * 16 + k2];
  }
  __syncthreads();
  for (int r = 0; r < 64; ++r) {
    int src = sIdx[r];
    sX[r * kLDA + tid] = (f16)A.f2[((size_t)b * 8192 + src) * 256 + tid];
  }
  __syncthreads();

  const int w = tid >> 5, lane = tid & 31, kh = lane >> 4, ln = lane & 15;
  const int mt = w & 3, wg = w >> 2, M0 = mt * 16;

  Frag a[8];
#pragma unroll
  for (int s = 0; s < 8; ++s) {
    const f16* p = sX + (M0 + ln) * kLDA + s * 32 + kh * 8;
    a[s].q[0] = *(const u32x4*)p;
    a[s].q[1] = *(const u32x4*)(p + 16);
  }
#pragma unroll 1
  for (int j = 0; j < 16; ++j) {           // 32 col tiles / 2 wave-groups
    const int nt = wg + 2 * j;
    const int N  = nt * 16 + ln;

    // BN affine operands issued before the WMMA chain (overlap)
    const float bv = A.bnv[N], bg = A.bng[N], bb2 = A.bnb[N];
    const float tb = A.tdb[N], bm = A.bnm[N];

    f32x8 acc = {};
    const f16* bp = A.tdwT + (size_t)N * 256 + kh * 8;
#pragma unroll
    for (int s = 0; s < 8; ++s) {
      Frag bb;
      bb.q[0] = *(const u32x4*)bp;
      bb.q[1] = *(const u32x4*)(bp + 16);
      bp += 32;
      acc = wmma_step(a[s], bb, acc);
    }
    const float rs = rsqrtf(bv + 1e-5f);
    const float sc = bg * rs;
    const float sh = bb2 + (tb - bm) * sc;
    float m = 0.0f;                          // relu floor; max(relu(x)) == relu(max(x))
#pragma unroll
    for (int r = 0; r < 8; ++r) m = fmaxf(m, acc[r] * sc + sh);
    m = fmaxf(m, __shfl_xor(m, 16));         // combine the two half-wave row groups
    if (kh == 0)
      A.outF[((size_t)(b * 2048 + p0 + mt)) * 512 + N] = m;
  }
}

// ---------------------------------------------------------------------------
// pos_ds = pos gathered by fps_idx
// ---------------------------------------------------------------------------
__global__ void gather_pos_kernel(const float* __restrict__ pos,
                                  const int* __restrict__ fps,
                                  float* __restrict__ outp) {
  int i = blockIdx.x * 256 + threadIdx.x;   // over B*M = 16384
  if (i < 8 * 2048) {
    int b = i >> 11;
    int s = fps[i];
    const float* pp = pos + ((size_t)b * 8192 + s) * 3;
    outp[i * 3 + 0] = pp[0];
    outp[i * 3 + 1] = pp[1];
    outp[i * 3 + 2] = pp[2];
  }
}

// ---------------------------------------------------------------------------
extern "C" void kernel_launch(void* const* d_in, const int* in_sizes, int n_in,
                              void* d_out, int out_size, void* d_ws, size_t ws_size,
                              hipStream_t stream) {
  (void)in_sizes; (void)n_in; (void)out_size; (void)ws_size;

  const float* pos  = (const float*)d_in[0];
  const float* feat = (const float*)d_in[1];
  const int*   fps  = (const int*)d_in[2];
  const int*   knn  = (const int*)d_in[3];
  const float* m1w1 = (const float*)d_in[6];
  const float* m1b1 = (const float*)d_in[7];
  const float* m1w2 = (const float*)d_in[8];
  const float* m1b2 = (const float*)d_in[9];
  const float* m2w1 = (const float*)d_in[10];
  const float* m2b1 = (const float*)d_in[11];
  const float* m2w2 = (const float*)d_in[12];
  const float* m2b2 = (const float*)d_in[13];
  const float* wq   = (const float*)d_in[14];
  const float* wk   = (const float*)d_in[15];
  const float* wv   = (const float*)d_in[16];
  const float* wo   = (const float*)d_in[17];
  const float* bo   = (const float*)d_in[18];
  const float* lng  = (const float*)d_in[19];
  const float* lnb  = (const float*)d_in[20];
  const float* tdw  = (const float*)d_in[21];
  const float* tdb  = (const float*)d_in[22];
  const float* bng  = (const float*)d_in[23];
  const float* bnb  = (const float*)d_in[24];
  const float* bnm  = (const float*)d_in[25];
  const float* bnv  = (const float*)d_in[26];

  // workspace: [0,2MB) f16 weights | f1 64MB | f2 64MB
  f16*   wsW = (f16*)d_ws;
  float* f1  = (float*)((char*)d_ws + (size_t)2 * 1024 * 1024);
  float* f2  = f1 + (size_t)8 * 8192 * 256;

  auto cvt = [&](const float* s, f16* dst, int K, int N) {
    int total = K * N;
    convT_kernel<<<(total + 255) / 256, 256, 0, stream>>>(s, dst, K, N);
  };
  for (int i = 0; i < 2; ++i) {
    const float* mats[6] = { m1w2 + (size_t)i * 65536, m2w2 + (size_t)i * 65536,
                             wq + (size_t)i * 65536,   wk + (size_t)i * 65536,
                             wv + (size_t)i * 65536,   wo + (size_t)i * 65536 };
    for (int m = 0; m < 6; ++m)
      cvt(mats[m], wsW + ((size_t)i * 6 + m) * 65536, 256, 256);
  }
  f16* tdwT = wsW + (size_t)12 * 65536;
  cvt(tdw, tdwT, 256, 512);

  for (int i = 0; i < 2; ++i) {
    BlockArgs ba;
    ba.pos  = pos;
    ba.fin  = (i == 0) ? feat : f1;
    ba.fout = (i == 0) ? f1 : f2;
    const f16* base = wsW + (size_t)i * 6 * 65536;
    ba.m1w2t = base;             ba.m2w2t = base + 65536;
    ba.wqt   = base + 2 * 65536; ba.wkt   = base + 3 * 65536;
    ba.wvt   = base + 4 * 65536; ba.wot   = base + 5 * 65536;
    ba.m1w1 = m1w1 + (size_t)i * 1024; ba.m1b1 = m1b1 + (size_t)i * 256;
    ba.m1b2 = m1b2 + (size_t)i * 256;
    ba.m2w1 = m2w1 + (size_t)i * 1536; ba.m2b1 = m2b1 + (size_t)i * 256;
    ba.m2b2 = m2b2 + (size_t)i * 256;
    ba.bo = bo + (size_t)i * 256; ba.lng = lng + (size_t)i * 256; ba.lnb = lnb + (size_t)i * 256;
    ba.cs = (i == 0) ? 16 : 64;
    block_kernel<<<1024, 256, SMEM_BYTES, stream>>>(ba);
  }

  DownArgs da;
  da.f2 = f2; da.knn = knn; da.tdwT = tdwT;
  da.tdb = tdb; da.bng = bng; da.bnb = bnb; da.bnm = bnm; da.bnv = bnv;
  da.outF = (float*)d_out + (size_t)8 * 2048 * 3;
  down_kernel<<<4096, 256, 0, stream>>>(da);

  gather_pos_kernel<<<(8 * 2048 + 255) / 256, 256, 0, stream>>>(pos, fps, (float*)d_out);
}
